// L2TransformerBlock_38190849196710
// MI455X (gfx1250) — matrix operand order repaired and checked by reading, not compiled, not run on hardware
//
#include <hip/hip_runtime.h>
#include <hip/hip_bf16.h>

// B=2, S=2048, D=1024, H=16, DK=64  -> BS = B*S = 4096, BH = 32
#define Bz 2
#define Sz 2048
#define Dz 1024
#define Hz 16
#define DKz 64
#define BSz (Bz*Sz)
#define BHz (Bz*Hz)
#define C1c (23.0f/24.0f)
#define C2c (1.0f/24.0f)
#define RSCALE 0.125f           // 1/sqrt(DK)
#define GELU_RDIV (1.0f/1.1289f)

typedef __attribute__((ext_vector_type(16))) __bf16 v16bf;
typedef __attribute__((ext_vector_type(8)))  float  v8f;
typedef unsigned int u32x4 __attribute__((ext_vector_type(4)));
typedef int          i32x4 __attribute__((ext_vector_type(4)));
typedef int          i32x8 __attribute__((ext_vector_type(8)));

union U16B { uint4 u; __bf16 h[8]; };

#if __has_builtin(__builtin_amdgcn_tensor_load_to_lds)
#define HAVE_TDM 1
#else
#define HAVE_TDM 0
#endif

__device__ __forceinline__ v8f wmma_bf16(v16bf a, v16bf b, v8f c) {
  return __builtin_amdgcn_wmma_f32_16x16x32_bf16(false, a, false, b, (short)0, c, false, false);
}
// 16-bit A-matrix 16x32 K index for element e (=2v+j), lane-half lh
__device__ __forceinline__ int a_kk(int e, int lh) {
  return (e < 8) ? (lh * 8 + e) : (16 + lh * 8 + (e - 8));
}
// 16-bit B-matrix 32x16 K index
__device__ __forceinline__ int b_kk(int e, int lh) { return (e < 8) ? (lh * 8 + e) : (16 + lh * 8 + (e - 8)); }

// LDS byte offset: ISA 10.2 -- LDS aperture is in addr[63:32]; addr[31:0] is the offset.
__device__ __forceinline__ unsigned lds_off_of(const void* p) {
  return (unsigned)(unsigned long long)p;
}

__device__ __forceinline__ void wait_tensor0() {
#if __has_builtin(__builtin_amdgcn_s_wait_tensorcnt)
  __builtin_amdgcn_s_wait_tensorcnt(0);
#else
  asm volatile("s_wait_tensorcnt 0x0" ::: "memory");
#endif
}

#if HAVE_TDM
// One 2D tile: rows of td0_units * 8 bytes, td1_rows rows, global row stride
// stride_units * 8 bytes, LDS padded by (pad_amount+1) DWORDs every
// 2^(pad_interval+1) DWORDs.  data_size = 8B.
__device__ __forceinline__ void tdm_load_2d(unsigned lds_byte_off, const void* gptr,
                                            unsigned td0_units, unsigned td1_rows,
                                            unsigned stride_units,
                                            unsigned pad_interval, unsigned pad_amount) {
  unsigned long long ga = (unsigned long long)gptr;
  u32x4 g0;
  g0[0] = 1u;                                      // count=1, user mode
  g0[1] = lds_byte_off;                            // lds_addr
  g0[2] = (unsigned)ga;                            // global_addr[31:0]
  g0[3] = (unsigned)((ga >> 32) & 0x1ffffffull) | (2u << 30); // addr[56:32] | type=2
  const unsigned dim0 = 1u << 20, dim1 = 1u << 20; // generous OOB bounds
  i32x8 g1;
  g1[0] = (int)((3u << 16) | (1u << 20) | (pad_interval << 22) | (pad_amount << 25));
  g1[1] = (int)((dim0 & 0xffffu) << 16);
  g1[2] = (int)((dim0 >> 16) | ((dim1 & 0xffffu) << 16));
  g1[3] = (int)((dim1 >> 16) | (td0_units << 16));
  g1[4] = (int)(td1_rows & 0xffffu);               // tile_dim1 (tile_dim2=0)
  g1[5] = (int)stride_units;                       // tensor_dim0_stride[31:0]
  g1[6] = 0;
  g1[7] = 0;
  i32x4 z4 = {0, 0, 0, 0};
#if defined(__clang_major__) && __clang_major__ >= 23
  i32x8 z8 = {0, 0, 0, 0, 0, 0, 0, 0};
  __builtin_amdgcn_tensor_load_to_lds(g0, g1, z4, z4, z8, 0);
#else
  __builtin_amdgcn_tensor_load_to_lds(g0, g1, z4, z4, 0);
#endif
}
#endif

// ---------------------------------------------------------------- converts
__global__ void f2bf_k(const float* __restrict__ in, __bf16* __restrict__ out, int n) {
  int i = blockIdx.x * blockDim.x + threadIdx.x;
  if (i < n) out[i] = (__bf16)in[i];
}

// ---------------------------------------------------------------- WMMA GEMM
// C[M,N] = A[M,K] @ W[N,K]^T, A/W bf16, accum f32.  BM=BN=128, BK=32.
// Double-buffered LDS; TDM prefetch of tile k+1 overlaps WMMA on tile k.
// EPI 0: outF = C
// EPI 1: r = C1*aux + C2*C ; outF = r ; outB = bf16(r)
// EPI 2: outB = bf16(gelu_exact(C)/1.1289)
// EPI 3: outF = C1*aux + C2*C
template <int EPI>
__global__ __launch_bounds__(256) void gemm_bf16_k(
    const __bf16* __restrict__ A, const __bf16* __restrict__ W,
    int M, int N, int K,
    float* __restrict__ outF, __bf16* __restrict__ outB,
    const float* __restrict__ aux) {
  __shared__ __align__(16) __bf16 lA[2][128][40];   // 64B data + 16B pad per row
  __shared__ __align__(16) __bf16 lW[2][128][40];

  const int tid  = threadIdx.x;
  const int wave = tid >> 5;
  const int lane = tid & 31;
  const int lh   = lane >> 4;
  const int ln   = lane & 15;
  const int wm   = wave >> 2;   // 0..1 -> 64 rows each
  const int wn   = wave & 3;    // 0..3 -> 32 cols each
  const int bm0  = blockIdx.y * 128;
  const int bn0  = blockIdx.x * 128;
  const int ksteps = K >> 5;

  v8f acc[4][2];
#pragma unroll
  for (int i = 0; i < 4; ++i)
#pragma unroll
    for (int j = 0; j < 2; ++j)
#pragma unroll
      for (int v = 0; v < 8; ++v) acc[i][j][v] = 0.0f;

#if HAVE_TDM
  if (wave == 0) {   // prologue: tile 0 via Tensor Data Mover
    tdm_load_2d(lds_off_of(&lA[0][0][0]), A + (size_t)bm0 * K, 8, 128, (unsigned)(K >> 2), 3, 3);
    tdm_load_2d(lds_off_of(&lW[0][0][0]), W + (size_t)bn0 * K, 8, 128, (unsigned)(K >> 2), 3, 3);
  }
#else
  {
#pragma unroll
    for (int i = 0; i < 2; ++i) {
      int c = tid + 256 * i;
      int r = c >> 2, co = (c & 3) * 8;
      U16B ta, tw;
      ta.u = *(const uint4*)(A + (size_t)(bm0 + r) * K + co);
      tw.u = *(const uint4*)(W + (size_t)(bn0 + r) * K + co);
#pragma unroll
      for (int j = 0; j < 8; ++j) { lA[0][r][co + j] = ta.h[j]; lW[0][r][co + j] = tw.h[j]; }
    }
  }
#endif

  for (int ks = 0; ks < ksteps; ++ks) {
    const int cb = ks & 1;
#if HAVE_TDM
    if (wave == 0) wait_tensor0();
#endif
    __syncthreads();   // tile ks visible; everyone done reading buffer cb^1

    if (ks + 1 < ksteps) {
      const int nb = cb ^ 1;
      const int k1 = (ks + 1) << 5;
#if HAVE_TDM
      if (wave == 0) {
        tdm_load_2d(lds_off_of(&lA[nb][0][0]), A + (size_t)bm0 * K + k1, 8, 128, (unsigned)(K >> 2), 3, 3);
        tdm_load_2d(lds_off_of(&lW[nb][0][0]), W + (size_t)bn0 * K + k1, 8, 128, (unsigned)(K >> 2), 3, 3);
      }
#else
#pragma unroll
      for (int i = 0; i < 2; ++i) {
        int c = tid + 256 * i;
        int r = c >> 2, co = (c & 3) * 8;
        U16B ta, tw;
        ta.u = *(const uint4*)(A + (size_t)(bm0 + r) * K + k1 + co);
        tw.u = *(const uint4*)(W + (size_t)(bn0 + r) * K + k1 + co);
#pragma unroll
        for (int j = 0; j < 8; ++j) { lA[nb][r][co + j] = ta.h[j]; lW[nb][r][co + j] = tw.h[j]; }
      }
#endif
    }

    v16bf af[4], bf[2];
#pragma unroll
    for (int mt = 0; mt < 4; ++mt) {
      int row = wm * 64 + mt * 16 + ln;
#pragma unroll
      for (int e = 0; e < 16; ++e) af[mt][e] = lA[cb][row][a_kk(e, lh)];
    }
#pragma unroll
    for (int nt = 0; nt < 2; ++nt) {
      int col = wn * 32 + nt * 16 + ln;
#pragma unroll
      for (int e = 0; e < 16; ++e) bf[nt][e] = lW[cb][col][b_kk(e, lh)];
    }
#pragma unroll
    for (int mt = 0; mt < 4; ++mt)
#pragma unroll
      for (int nt = 0; nt < 2; ++nt)
        acc[mt][nt] = wmma_bf16(af[mt], bf[nt], acc[mt][nt]);
  }

#pragma unroll
  for (int mt = 0; mt < 4; ++mt)
#pragma unroll
    for (int nt = 0; nt < 2; ++nt) {
      int col = bn0 + wn * 32 + nt * 16 + ln;
#pragma unroll
      for (int v = 0; v < 8; ++v) {
        int row = bm0 + wm * 64 + mt * 16 + v + 8 * lh;
        size_t idx = (size_t)row * N + col;
        float c = acc[mt][nt][v];
        if (EPI == 0) {
          outF[idx] = c;
        } else if (EPI == 1) {
          float r = C1c * aux[idx] + C2c * c;
          outF[idx] = r;
          outB[idx] = (__bf16)r;
        } else if (EPI == 2) {
          float g = 0.5f * c * (1.0f + erff(c * 0.70710678118654752f)) * GELU_RDIV;
          outB[idx] = (__bf16)g;
        } else {
          outF[idx] = C1c * aux[idx] + C2c * c;
        }
      }
    }
}

// ------------------------------------------------- pack Q -> bf16 heads + qn
__global__ __launch_bounds__(256) void pack_q_k(const float* __restrict__ Y,
                                                __bf16* __restrict__ Qb,
                                                float* __restrict__ qn) {
  int row = blockIdx.x;                 // b*S + s
  int b = row / Sz, s = row % Sz;
  int t = threadIdx.x;                  // 256 threads, 4 elems each
  int h = t >> 4;
  int dk0 = (t & 15) * 4;
  const float* yr = Y + (size_t)row * Dz + t * 4;
  __bf16* qr = Qb + ((size_t)(b * Hz + h) * Sz + s) * DKz + dk0;
  float ss = 0.0f;
#pragma unroll
  for (int j = 0; j < 4; ++j) {
    float v = yr[j];
    ss += v * v;
    qr[j] = (__bf16)v;
  }
#pragma unroll
  for (int m = 1; m < 16; m <<= 1) ss += __shfl_xor(ss, m, 32);
  if ((t & 15) == 0) qn[(size_t)(b * Hz + h) * Sz + s] = ss;
}

// ------------------------------------------------- merged[h,k,v] (w_k.w_v^T)/8
__global__ __launch_bounds__(64) void merged_k(const float* __restrict__ wk,
                                               const float* __restrict__ wv,
                                               float* __restrict__ merged) {
  int hk = blockIdx.x;                  // h*64 + k
  __shared__ float wrow[Dz];
#pragma unroll
  for (int i = 0; i < 16; ++i) wrow[threadIdx.x + 64 * i] = wk[(size_t)hk * Dz + threadIdx.x + 64 * i];
  __syncthreads();
  int h = hk >> 6, v = threadIdx.x;
  const float* wvr = wv + (size_t)(h * 64 + v) * Dz;
  float acc = 0.0f;
  for (int d = 0; d < Dz; ++d) acc += wrow[d] * wvr[d];
  merged[(size_t)hk * 64 + v] = acc * RSCALE;   // fold /sqrt(dk)
}

// ------------------------------------------------- flash attention (L2 scores)
// grid: (S/64, B*H), block 128 (4 waves x 16 q-rows). K tiles via TDM.
__global__ __launch_bounds__(128) void attn_k(const __bf16* __restrict__ Qb,
                                              const float* __restrict__ qn,
                                              float* __restrict__ ctx) {
  const int qt = blockIdx.x;
  const int bh = blockIdx.y;
  const int tid  = threadIdx.x;
  const int wave = tid >> 5;
  const int lane = tid & 31;
  const int lh   = lane >> 4;
  const int ln   = lane & 15;

  __shared__ __align__(16) __bf16 lQ[64][72];   // 128B data + 16B pad per row
  __shared__ __align__(16) __bf16 lK[64][72];
  __shared__ __align__(16) __bf16 lP[4][16][72];
  __shared__ float  lkn[64];

  // load q tile (64 x 64)
  const __bf16* qbase = Qb + ((size_t)bh * Sz + qt * 64) * DKz;
#if HAVE_TDM
  if (wave == 0) {
    tdm_load_2d(lds_off_of(&lQ[0][0]), qbase, 16, 64, 16, 4, 3);
    wait_tensor0();
  }
#else
#pragma unroll
  for (int i = 0; i < 4; ++i) {
    int c = tid + 128 * i;
    int r = c >> 3, co = (c & 7) * 8;
    U16B tq; tq.u = *(const uint4*)(qbase + (size_t)r * DKz + co);
#pragma unroll
    for (int j = 0; j < 8; ++j) lQ[r][co + j] = tq.h[j];
  }
#endif
  __syncthreads();

  // per-wave Q A-fragments (kept for whole kernel)
  v16bf aq[2];
#pragma unroll
  for (int f = 0; f < 2; ++f)
#pragma unroll
    for (int e = 0; e < 16; ++e) aq[f][e] = lQ[wave * 16 + ln][f * 32 + a_kk(e, lh)];

  float qn_r[8];
#pragma unroll
  for (int v = 0; v < 8; ++v)
    qn_r[v] = qn[(size_t)bh * Sz + qt * 64 + wave * 16 + v + 8 * lh];

  v8f o[4];
  float m_i[8], l_i[8];
#pragma unroll
  for (int t = 0; t < 4; ++t)
#pragma unroll
    for (int v = 0; v < 8; ++v) o[t][v] = 0.0f;
#pragma unroll
  for (int v = 0; v < 8; ++v) { m_i[v] = -3.0e38f; l_i[v] = 0.0f; }

  for (int kt = 0; kt <= qt; ++kt) {
    __syncthreads();
    const __bf16* kbase = Qb + ((size_t)bh * Sz + kt * 64) * DKz;
#if HAVE_TDM
    if (wave == 0) tdm_load_2d(lds_off_of(&lK[0][0]), kbase, 16, 64, 16, 4, 3);
#else
#pragma unroll
    for (int i = 0; i < 4; ++i) {
      int c = tid + 128 * i;
      int r = c >> 3, co = (c & 7) * 8;
      U16B tk; tk.u = *(const uint4*)(kbase + (size_t)r * DKz + co);
#pragma unroll
      for (int j = 0; j < 8; ++j) lK[r][co + j] = tk.h[j];
    }
#endif
    if (tid < 64) lkn[tid] = qn[(size_t)bh * Sz + kt * 64 + tid];
#if HAVE_TDM
    if (wave == 0) wait_tensor0();
#endif
    __syncthreads();

    // scores: -(qn - 2 Q.K + kn)/8 with causal mask
    v8f sc[4];
#pragma unroll
    for (int t = 0; t < 4; ++t) {
      v8f dot;
#pragma unroll
      for (int v = 0; v < 8; ++v) dot[v] = 0.0f;
#pragma unroll
      for (int f = 0; f < 2; ++f) {
        v16bf bk;
#pragma unroll
        for (int e = 0; e < 16; ++e) bk[e] = lK[16 * t + ln][f * 32 + b_kk(e, lh)];
        dot = wmma_bf16(aq[f], bk, dot);
      }
      float knv = lkn[16 * t + ln];
      int key = kt * 64 + 16 * t + ln;
#pragma unroll
      for (int v = 0; v < 8; ++v) {
        int row = qt * 64 + wave * 16 + v + 8 * lh;
        float s = -(qn_r[v] - 2.0f * dot[v] + knv) * RSCALE;
        if (key > row) s = -1.0e9f;
        sc[t][v] = s;
      }
    }

    // online softmax update (row reductions across 16 lanes)
#pragma unroll
    for (int v = 0; v < 8; ++v) {
      float mx = fmaxf(fmaxf(sc[0][v], sc[1][v]), fmaxf(sc[2][v], sc[3][v]));
#pragma unroll
      for (int m = 1; m < 16; m <<= 1) mx = fmaxf(mx, __shfl_xor(mx, m, 32));
      float mn = fmaxf(m_i[v], mx);
      float alpha = __expf(m_i[v] - mn);
      float rs = 0.0f;
#pragma unroll
      for (int t = 0; t < 4; ++t) {
        float p = __expf(sc[t][v] - mn);
        sc[t][v] = p;
        rs += p;
      }
#pragma unroll
      for (int m = 1; m < 16; m <<= 1) rs += __shfl_xor(rs, m, 32);
      l_i[v] = l_i[v] * alpha + rs;
      m_i[v] = mn;
#pragma unroll
      for (int t = 0; t < 4; ++t) o[t][v] *= alpha;
    }

    // round-trip P through per-wave LDS: C-layout -> A-layout
#pragma unroll
    for (int t = 0; t < 4; ++t)
#pragma unroll
      for (int v = 0; v < 8; ++v)
        lP[wave][v + 8 * lh][16 * t + ln] = (__bf16)sc[t][v];
    asm volatile("s_wait_dscnt 0x0" ::: "memory");

    v16bf pa[2];
#pragma unroll
    for (int f = 0; f < 2; ++f)
#pragma unroll
      for (int e = 0; e < 16; ++e) pa[f][e] = lP[wave][ln][f * 32 + a_kk(e, lh)];

    // O += P(16x64) @ K(64x64)
#pragma unroll
    for (int tt = 0; tt < 4; ++tt) {
#pragma unroll
      for (int f = 0; f < 2; ++f) {
        v16bf bo;
#pragma unroll
        for (int e = 0; e < 16; ++e) bo[e] = lK[f * 32 + b_kk(e, lh)][16 * tt + ln];
        o[tt] = wmma_bf16(pa[f], bo, o[tt]);
      }
    }
  }

  // normalize, store context f32 [BH, S, DK]
#pragma unroll
  for (int v = 0; v < 8; ++v) {
    float rl = 1.0f / l_i[v];
    int row = qt * 64 + wave * 16 + v + 8 * lh;
#pragma unroll
    for (int t = 0; t < 4; ++t)
      ctx[((size_t)bh * Sz + row) * DKz + 16 * t + ln] = o[t][v] * rl;
  }
}

// ------------------------------------------------- ctx @ merged -> bf16 [BS,D]
__global__ __launch_bounds__(256) void ctx2_k(const float* __restrict__ ctx,
                                              const float* __restrict__ merged,
                                              __bf16* __restrict__ out) {
  int row = blockIdx.x;                 // b*S + s
  int b = row / Sz, s = row % Sz;
  __shared__ float cr[Dz];
#pragma unroll
  for (int i = 0; i < 4; ++i) {
    int idx = threadIdx.x + 256 * i;
    int h = idx >> 6, k = idx & 63;
    cr[idx] = ctx[((size_t)(b * Hz + h) * Sz + s) * DKz + k];
  }
  __syncthreads();
#pragma unroll
  for (int i = 0; i < 4; ++i) {
    int col = threadIdx.x + 256 * i;
    int h = col >> 6, v = col & 63;
    float acc = 0.0f;
    for (int k = 0; k < 64; ++k) acc += cr[h * 64 + k] * merged[(size_t)(h * 64 + k) * 64 + v];
    out[(size_t)row * Dz + col] = (__bf16)acc;
  }
}

// =============================================================== launch
extern "C" void kernel_launch(void* const* d_in, const int* in_sizes, int n_in,
                              void* d_out, int out_size, void* d_ws, size_t ws_size,
                              hipStream_t stream) {
  const float* x     = (const float*)d_in[0];
  // d_in[1] = mask (tril) -- causal mask is applied analytically
  const float* w_k   = (const float*)d_in[2];
  const float* w_v   = (const float*)d_in[3];
  const float* w_o   = (const float*)d_in[4];
  const float* w_in  = (const float*)d_in[5];
  const float* w_out = (const float*)d_in[6];
  float* outp = (float*)d_out;

  char* ws = (char*)d_ws;
  size_t off = 0;
  auto alloc = [&](size_t bytes) { size_t o = off; off += (bytes + 255) & ~(size_t)255; return o; };

  // 16MB float region reused: Yf (Q-proj out) -> ctx -> x1f
  float*  fbuf  = (float*)(ws + alloc((size_t)BSz * Dz * 4));
  // 8MB bf16 region reused: xb -> ctx2b
  __bf16* bf8   = (__bf16*)(ws + alloc((size_t)BSz * Dz * 2));
  __bf16* wkb   = (__bf16*)(ws + alloc((size_t)Dz * Dz * 2));
  __bf16* wob   = (__bf16*)(ws + alloc((size_t)Dz * Dz * 2));
  __bf16* winb  = (__bf16*)(ws + alloc((size_t)4 * Dz * Dz * 2));
  __bf16* woutb = (__bf16*)(ws + alloc((size_t)4 * Dz * Dz * 2));
  __bf16* Qb    = (__bf16*)(ws + alloc((size_t)BHz * Sz * DKz * 2));
  float*  qn    = (float*)(ws + alloc((size_t)BHz * Sz * 4));
  float*  mrg   = (float*)(ws + alloc((size_t)Hz * DKz * DKz * 4));
  __bf16* x1b   = (__bf16*)(ws + alloc((size_t)BSz * Dz * 2));
  __bf16* hb    = (__bf16*)(ws + alloc((size_t)BSz * 4 * Dz * 2));

  // 1) bf16 converts
  int nX = BSz * Dz, nW = Dz * Dz, nWI = 4 * Dz * Dz;
  f2bf_k<<<(nX + 255) / 256, 256, 0, stream>>>(x, bf8, nX);
  f2bf_k<<<(nW + 255) / 256, 256, 0, stream>>>(w_k, wkb, nW);
  f2bf_k<<<(nW + 255) / 256, 256, 0, stream>>>(w_o, wob, nW);
  f2bf_k<<<(nWI + 255) / 256, 256, 0, stream>>>(w_in, winb, nWI);
  f2bf_k<<<(nWI + 255) / 256, 256, 0, stream>>>(w_out, woutb, nWI);

  // 2) Q = x @ w_k^T  (f32 out in fbuf)
  gemm_bf16_k<0><<<dim3(Dz / 128, BSz / 128), 256, 0, stream>>>(
      bf8, wkb, BSz, Dz, Dz, fbuf, nullptr, nullptr);

  // 3) pack per-head bf16 Q + row norms
  pack_q_k<<<BSz, 256, 0, stream>>>(fbuf, Qb, qn);

  // 4) merged = (w_k_h . w_v_h^T)/sqrt(dk)
  merged_k<<<Hz * DKz, 64, 0, stream>>>(w_k, w_v, mrg);

  // 5) flash attention, ctx (f32) into fbuf (Yf dead)
  attn_k<<<dim3(Sz / 64, BHz), 128, 0, stream>>>(Qb, qn, fbuf);

  // 6) ctx @ merged -> bf16 [BS, D] into bf8 (xb dead)
  ctx2_k<<<BSz, 256, 0, stream>>>(fbuf, mrg, bf8);

  // 7) attn_out @ w_o^T, residual: x1 = c1*x + c2*ao ; fbuf := x1f, x1b bf16
  gemm_bf16_k<1><<<dim3(Dz / 128, BSz / 128), 256, 0, stream>>>(
      bf8, wob, BSz, Dz, Dz, fbuf, x1b, x);

  // 8) h = gelu(x1 @ w_in^T)/1.1289 -> bf16
  gemm_bf16_k<2><<<dim3(4 * Dz / 128, BSz / 128), 256, 0, stream>>>(
      x1b, winb, BSz, 4 * Dz, Dz, nullptr, hb, nullptr);

  // 9) out = c1*x1 + c2*(h @ w_out^T)
  gemm_bf16_k<3><<<dim3(Dz / 128, BSz / 128), 256, 0, stream>>>(
      hb, woutb, BSz, Dz, 4 * Dz, outp, nullptr, fbuf);
}
